// GraphTransformerLayer_33011118637080
// MI455X (gfx1250) — matrix operand order repaired and checked by reading, not compiled, not run on hardware
//
#include <hip/hip_runtime.h>
#include <hip/hip_bf16.h>
#include <stdint.h>

// ---------------------------------------------------------------------------
// CDNA5 (gfx1250, wave32) types and feature gates
// ---------------------------------------------------------------------------
typedef __attribute__((ext_vector_type(16))) __bf16    v16bf;
typedef __attribute__((ext_vector_type(8)))  float     v8f;
typedef __attribute__((ext_vector_type(8)))  uint16_t  u16x8;
typedef __attribute__((ext_vector_type(4)))  int       v4i;

#if defined(__has_builtin)
#  if __has_builtin(__builtin_amdgcn_global_load_async_to_lds_b128)
#    define HAVE_ASYNC_LDS 1
#  endif
#  if __has_builtin(__builtin_amdgcn_s_wait_asynccnt)
#    define HAVE_WAIT_ASYNC 1
#  endif
#endif
#ifndef HAVE_ASYNC_LDS
#  define HAVE_ASYNC_LDS 0
#endif
#ifndef HAVE_WAIT_ASYNC
#  define HAVE_WAIT_ASYNC 0
#endif

static __device__ __forceinline__ void wait_async0() {
#if HAVE_WAIT_ASYNC
    __builtin_amdgcn_s_wait_asynccnt(0);
#elif HAVE_ASYNC_LDS
    asm volatile("s_wait_asynccnt 0" ::: "memory");
#endif
}

static __device__ __forceinline__ uint16_t f2bf_bits(float f) {
    union { float f; uint32_t u; } x; x.f = f;
    uint32_t u = x.u;
    uint32_t r = u + 0x7FFFu + ((u >> 16) & 1u);   // round-to-nearest-even
    return (uint16_t)(r >> 16);
}

// ---------------------------------------------------------------------------
// bf16 WMMA GEMM:  out[M,N] = A[M,K] @ B[N,K]^T + bias[N]
// Block = 8 waves covering a 16(M) x 256(N) strip; each wave owns a 16x32 tile
// (one A fragment, two B fragments, two accumulators per K-step).
// The 16-row A strip is staged once per block into LDS via async-to-LDS
// (GLOBAL_LOAD_ASYNC_TO_LDS_B128, ASYNCcnt) to kill the 8x redundant A reads.
// Fragment layouts used here (wave32; A and B share contiguous k order):
//   A 16x32 bf16 : row = lane&15 ; elems 0..7 -> K = koffA+0..7,
//                  elems 8..15 -> K = koffA+8..15, koffA = (lane>>4)*16
//   B 32x16 bf16 : col = lane&15 ; elem i -> K = (lane>>4)*16 + i
//   C/D f32      : elem r -> (m = r + (lane>>4)*8, n = lane&15)
// ---------------------------------------------------------------------------
__global__ void __launch_bounds__(256)
gemm_bf16_wmma(const uint16_t* __restrict__ A,
               const uint16_t* __restrict__ B,
               const float*    __restrict__ bias,
               const float*    __restrict__ resid,
               const float*    __restrict__ alpha_p,
               float*          __restrict__ outF,
               uint16_t*       __restrict__ outBF,
               int M, int N, int K)
{
    extern __shared__ uint16_t sA[];         // 16 x K bf16 (8KB @K=256, 32KB @K=1024)
    const int tid  = threadIdx.x;
    const int lane = tid & 31;
    const int wave = tid >> 5;
    const int m0   = blockIdx.x << 4;
    const int n0   = (blockIdx.y * 8 + wave) << 5;   // 32 output columns per wave

    // ---- stage A strip (16 rows x K) into LDS, once per block ----
    const uint16_t* Ablk = A + (size_t)m0 * K;
    const int chunks = (16 * K) >> 3;                 // 16-byte chunks
#if HAVE_ASYNC_LDS
    for (int c = tid; c < chunks; c += 256) {
        __builtin_amdgcn_global_load_async_to_lds_b128(
            (v4i*)(Ablk + c * 8), (v4i*)(sA + c * 8), 0, 0);
    }
    wait_async0();
#else
    for (int c = tid; c < chunks; c += 256) {
        *(u16x8*)(sA + c * 8) = *(const u16x8*)(Ablk + c * 8);
    }
#endif
    __syncthreads();

    const int koffA = (lane >> 4) << 4;    // 0 or 16
    const int koffB = (lane >> 4) << 4;    // 0 or 16
    const uint16_t* sAp = sA + (lane & 15) * K;
    const uint16_t* B0  = B + (size_t)(n0 + (lane & 15)) * K;
    const uint16_t* B1  = B0 + (size_t)16 * K;

    union Frag { u16x8 h[2]; v16bf v; };

    const float bn0 = bias[n0 + (lane & 15)];
    const float bn1 = bias[n0 + 16 + (lane & 15)];
    v8f acc0, acc1;
#pragma unroll
    for (int r = 0; r < 8; ++r) { acc0[r] = bn0; acc1[r] = bn1; }

    for (int kk = 0; kk < K; kk += 32) {
        Frag a, b0, b1;
        a.h[0]  = *(const u16x8*)(sAp + kk + koffA);        // ds_load_b128
        a.h[1]  = *(const u16x8*)(sAp + kk + koffA + 8);
        b0.h[0] = *(const u16x8*)(B0 + kk + koffB);
        b0.h[1] = *(const u16x8*)(B0 + kk + koffB + 8);
        b1.h[0] = *(const u16x8*)(B1 + kk + koffB);
        b1.h[1] = *(const u16x8*)(B1 + kk + koffB + 8);
        acc0 = __builtin_amdgcn_wmma_f32_16x16x32_bf16(
                   false, a.v, false, b0.v, (short)0, acc0, false, false);
        acc1 = __builtin_amdgcn_wmma_f32_16x16x32_bf16(
                   false, a.v, false, b1.v, (short)0, acc1, false, false);
    }

    const float alpha = (resid && alpha_p) ? alpha_p[0] : 1.0f;
    const int   nc0   = n0 + (lane & 15);
    const int   nc1   = nc0 + 16;
#pragma unroll
    for (int r = 0; r < 8; ++r) {
        const int m = m0 + r + ((lane >> 4) << 3);
        float v0 = acc0[r], v1 = acc1[r];
        if (resid) {
            v0 = resid[(size_t)m * N + nc0] + alpha * v0;
            v1 = resid[(size_t)m * N + nc1] + alpha * v1;
        }
        if (outBF) {  // exact-GELU epilogue -> bf16
            v0 = 0.5f * v0 * (1.0f + erff(v0 * 0.70710678118654752f));
            v1 = 0.5f * v1 * (1.0f + erff(v1 * 0.70710678118654752f));
            outBF[(size_t)m * N + nc0] = f2bf_bits(v0);
            outBF[(size_t)m * N + nc1] = f2bf_bits(v1);
        } else {
            outF[(size_t)m * N + nc0] = v0;
            outF[(size_t)m * N + nc1] = v1;
        }
    }
}

// ---------------------------------------------------------------------------
// LayerNorm over D=256, write bf16 activations. One block (8 waves) per row.
// ---------------------------------------------------------------------------
__global__ void ln_to_bf16(const float* __restrict__ x,
                           const float* __restrict__ g,
                           const float* __restrict__ b,
                           uint16_t*    __restrict__ out)
{
    const int row = blockIdx.x;
    const int tid = threadIdx.x;     // 256 threads
    const float v = x[(size_t)row * 256 + tid];

    __shared__ float s1[8], s2[8];
    float a = v, q = v * v;
#pragma unroll
    for (int off = 16; off; off >>= 1) {
        a += __shfl_xor(a, off, 32);
        q += __shfl_xor(q, off, 32);
    }
    if ((tid & 31) == 0) { s1[tid >> 5] = a; s2[tid >> 5] = q; }
    __syncthreads();
    float suma = 0.f, sumq = 0.f;
#pragma unroll
    for (int i = 0; i < 8; ++i) { suma += s1[i]; sumq += s2[i]; }
    const float mu  = suma * (1.0f / 256.0f);
    const float var = sumq * (1.0f / 256.0f) - mu * mu;
    const float inv = rsqrtf(var + 1e-5f);
    const float y = (v - mu) * inv * g[tid] + b[tid];
    out[(size_t)row * 256 + tid] = f2bf_bits(y);
}

__global__ void cvt_f32_bf16(const float* __restrict__ src,
                             uint16_t* __restrict__ dst, int n)
{
    int i = blockIdx.x * blockDim.x + threadIdx.x;
    if (i < n) dst[i] = f2bf_bits(src[i]);
}

__global__ void init_softmax(float* __restrict__ smax, float* __restrict__ ssum, int n)
{
    int i = blockIdx.x * blockDim.x + threadIdx.x;
    if (i < n) { smax[i] = -3.0e38f; ssum[i] = 0.0f; }
}

__global__ void zero_f32(float* __restrict__ p, int n)
{
    int i = blockIdx.x * blockDim.x + threadIdx.x;
    if (i < n) p[i] = 0.0f;
}

// ---------------------------------------------------------------------------
// Edge scores: one wave32 per edge; lane = d (HD=32).
// We (32x32 f32, 4KB) is staged once per block into LDS (async-to-LDS path).
// Fuses ep = edge_features @ We^T + be in-register.
// score[e,h] = (sum_d q[dst,h,d]*(k[src,h,d]+ep[d])) * scale * ew[e]
// ---------------------------------------------------------------------------
__global__ void __launch_bounds__(256)
edge_scores(const int*   __restrict__ ei,
            const float* __restrict__ ef,
            const float* __restrict__ ew,
            const float* __restrict__ We,
            const float* __restrict__ be,
            const float* __restrict__ q,
            const float* __restrict__ k,
            float* __restrict__ scores,
            float* __restrict__ smax,
            int E)
{
    __shared__ float sWe[1024];
    const int tid = threadIdx.x;
#if HAVE_ASYNC_LDS
    __builtin_amdgcn_global_load_async_to_lds_b128(
        (v4i*)(We + tid * 4), (v4i*)(&sWe[tid * 4]), 0, 0);
    wait_async0();
#else
    *(float4*)(&sWe[tid * 4]) = *(const float4*)(We + tid * 4);
#endif
    __syncthreads();

    const int lane = tid & 31;
    const int e = blockIdx.x * (blockDim.x >> 5) + (tid >> 5);
    if (e >= E) return;
    const int src = ei[e];
    const int dst = ei[E + e];

    const float myef = ef[(size_t)e * 32 + lane];
    float ep = be[lane];
#pragma unroll
    for (int j = 0; j < 32; ++j) {
        const float efj = __shfl(myef, j, 32);
        ep = fmaf(sWe[lane * 32 + j], efj, ep);
    }

    const float sw = 0.17677669529663688f * ew[e];  // 1/sqrt(32) * edge weight
    float myscore = 0.0f;
#pragma unroll
    for (int h = 0; h < 8; ++h) {
        const float kv = k[(size_t)src * 256 + h * 32 + lane] + ep;
        const float qv = q[(size_t)dst * 256 + h * 32 + lane];
        float p = qv * kv;
#pragma unroll
        for (int off = 16; off; off >>= 1) p += __shfl_xor(p, off, 32);
        if (lane == h) myscore = p * sw;
    }
    if (lane < 8) {
        scores[(size_t)e * 8 + lane] = myscore;
        atomicMax(&smax[(size_t)dst * 8 + lane], myscore);
    }
}

// exp(score - max) and segment-sum via atomics; thread per (edge,head)
__global__ void edge_exp(const int*   __restrict__ ei,
                         float*       __restrict__ scores,
                         const float* __restrict__ smax,
                         float*       __restrict__ ssum,
                         int E)
{
    int i = blockIdx.x * blockDim.x + threadIdx.x;
    if (i >= E * 8) return;
    const int e = i >> 3, h = i & 7;
    const int dst = ei[E + e];
    const float ex = expf(scores[i] - smax[(size_t)dst * 8 + h]);
    scores[i] = ex;
    atomicAdd(&ssum[(size_t)dst * 8 + h], ex);
}

// normalize, emit attn_weights, scatter w * v[src] into agg[dst]
__global__ void edge_aggregate(const int*   __restrict__ ei,
                               const float* __restrict__ scores,
                               const float* __restrict__ ssum,
                               const float* __restrict__ v,
                               float*       __restrict__ agg,
                               float*       __restrict__ attn_out,
                               int E)
{
    const int lane = threadIdx.x & 31;
    const int e = blockIdx.x * (blockDim.x >> 5) + (threadIdx.x >> 5);
    if (e >= E) return;
    const int src = ei[e], dst = ei[E + e];

    float w = 0.0f;
    if (lane < 8) {
        w = scores[(size_t)e * 8 + lane] / ssum[(size_t)dst * 8 + lane];
        attn_out[(size_t)e * 8 + lane] = w;
    }
#pragma unroll
    for (int h = 0; h < 8; ++h) {
        const float wh = __shfl(w, h, 32);
        atomicAdd(&agg[(size_t)dst * 256 + h * 32 + lane],
                  wh * v[(size_t)src * 256 + h * 32 + lane]);
    }
}

// ---------------------------------------------------------------------------
// Host orchestration
// ---------------------------------------------------------------------------
extern "C" void kernel_launch(void* const* d_in, const int* in_sizes, int n_in,
                              void* d_out, int out_size, void* d_ws, size_t ws_size,
                              hipStream_t stream)
{
    (void)n_in; (void)out_size; (void)ws_size;

    const float* x    = (const float*)d_in[0];
    const int*   ei   = (const int*)  d_in[1];
    const float* ef   = (const float*)d_in[2];
    const float* ew   = (const float*)d_in[3];
    const float* Wq   = (const float*)d_in[4];  const float* bq  = (const float*)d_in[5];
    const float* Wk   = (const float*)d_in[6];  const float* bk  = (const float*)d_in[7];
    const float* Wv   = (const float*)d_in[8];  const float* bv  = (const float*)d_in[9];
    const float* We   = (const float*)d_in[10]; const float* be_ = (const float*)d_in[11];
    const float* Wo   = (const float*)d_in[12]; const float* bo  = (const float*)d_in[13];
    const float* W1   = (const float*)d_in[14]; const float* b1  = (const float*)d_in[15];
    const float* W2   = (const float*)d_in[16]; const float* b2  = (const float*)d_in[17];
    const float* g1   = (const float*)d_in[18]; const float* be1 = (const float*)d_in[19];
    const float* g2   = (const float*)d_in[20]; const float* be2 = (const float*)d_in[21];
    const float* alpha= (const float*)d_in[22]; const float* beta= (const float*)d_in[23];

    const int N = in_sizes[0] / 256;   // 50000 (multiple of 16)
    const int E = in_sizes[3];         // 1600000

    // ---- workspace carve-out (256B aligned) ----
    char* ws = (char*)d_ws;
    size_t off = 0;
    auto alloc = [&](size_t bytes) -> char* {
        char* p = ws + off;
        off = (off + bytes + 255) & ~(size_t)255;
        return p;
    };
    float*    q      = (float*)   alloc((size_t)N * 256 * 4);   // later reused as x1
    float*    k      = (float*)   alloc((size_t)N * 256 * 4);   // later reused as agg
    float*    v      = (float*)   alloc((size_t)N * 256 * 4);
    float*    scores = (float*)   alloc((size_t)E * 8 * 4);
    float*    smax   = (float*)   alloc((size_t)N * 8 * 4);
    float*    ssum   = (float*)   alloc((size_t)N * 8 * 4);
    uint16_t* xnbf   = (uint16_t*)alloc((size_t)N * 256 * 2);   // later reused as agg_bf
    uint16_t* xn2bf  = (uint16_t*)alloc((size_t)N * 256 * 2);
    uint16_t* hbf    = (uint16_t*)alloc((size_t)N * 1024 * 2);
    uint16_t* wqb    = (uint16_t*)alloc(256 * 256 * 2);
    uint16_t* wkb    = (uint16_t*)alloc(256 * 256 * 2);
    uint16_t* wvb    = (uint16_t*)alloc(256 * 256 * 2);
    uint16_t* wob    = (uint16_t*)alloc(256 * 256 * 2);
    uint16_t* w1b    = (uint16_t*)alloc(1024 * 256 * 2);
    uint16_t* w2b    = (uint16_t*)alloc(256 * 1024 * 2);

    auto cdiv = [](long long a, long long b) { return (int)((a + b - 1) / b); };

    // 1) weights -> bf16
    cvt_f32_bf16<<<cdiv(256*256, 256), 256, 0, stream>>>(Wq, wqb, 256*256);
    cvt_f32_bf16<<<cdiv(256*256, 256), 256, 0, stream>>>(Wk, wkb, 256*256);
    cvt_f32_bf16<<<cdiv(256*256, 256), 256, 0, stream>>>(Wv, wvb, 256*256);
    cvt_f32_bf16<<<cdiv(256*256, 256), 256, 0, stream>>>(Wo, wob, 256*256);
    cvt_f32_bf16<<<cdiv(1024*256, 256), 256, 0, stream>>>(W1, w1b, 1024*256);
    cvt_f32_bf16<<<cdiv(256*1024, 256), 256, 0, stream>>>(W2, w2b, 256*1024);

    // 2) LN1 + QKV projections (WMMA, A strip staged via async-to-LDS)
    ln_to_bf16<<<N, 256, 0, stream>>>(x, g1, be1, xnbf);
    dim3 gD(N / 16, 1);                 // 8 waves x 32 cols = 256 output columns
    const int sh256  = 16 * 256 * 2;    // LDS bytes for K=256
    const int sh1024 = 16 * 1024 * 2;   // LDS bytes for K=1024
    gemm_bf16_wmma<<<gD, 256, sh256, stream>>>(xnbf, wqb, bq, nullptr, nullptr, q, nullptr, N, 256, 256);
    gemm_bf16_wmma<<<gD, 256, sh256, stream>>>(xnbf, wkb, bk, nullptr, nullptr, k, nullptr, N, 256, 256);
    gemm_bf16_wmma<<<gD, 256, sh256, stream>>>(xnbf, wvb, bv, nullptr, nullptr, v, nullptr, N, 256, 256);

    // 3) edge attention: scores + segment softmax + aggregate
    init_softmax<<<cdiv((long long)N * 8, 256), 256, 0, stream>>>(smax, ssum, N * 8);
    edge_scores<<<cdiv(E, 8), 256, 0, stream>>>(ei, ef, ew, We, be_, q, k, scores, smax, E);

    float* agg = k;  // k is dead after edge_scores
    zero_f32<<<cdiv((long long)N * 256, 256), 256, 0, stream>>>(agg, N * 256);
    edge_exp<<<cdiv((long long)E * 8, 256), 256, 0, stream>>>(ei, scores, smax, ssum, E);

    float* attnW = (float*)d_out + (size_t)N * 256;  // second output section [E,H]
    edge_aggregate<<<cdiv(E, 8), 256, 0, stream>>>(ei, scores, ssum, v, agg, attnW, E);

    // 4) Wo projection with fused residual: x1 = x + alpha * (agg @ Wo^T + bo)
    uint16_t* aggbf = xnbf;  // xnbf dead after QKV GEMMs
    cvt_f32_bf16<<<cdiv((long long)N * 256, 256), 256, 0, stream>>>(agg, aggbf, N * 256);
    float* x1 = q;           // q dead after edge_scores
    gemm_bf16_wmma<<<gD, 256, sh256, stream>>>(aggbf, wob, bo, x, alpha, x1, nullptr, N, 256, 256);

    // 5) FFN: LN2 -> GELU(x1@W1^T+b1) -> @W2^T+b2, residual with beta -> d_out
    ln_to_bf16<<<N, 256, 0, stream>>>(x1, g2, be2, xn2bf);
    dim3 gF(N / 16, 4);                 // 4 * 8 waves * 32 cols = 1024 output columns
    gemm_bf16_wmma<<<gF, 256, sh256, stream>>>(xn2bf, w1b, b1, nullptr, nullptr, nullptr, hbf, N, 1024, 256);
    gemm_bf16_wmma<<<gD, 256, sh1024, stream>>>(hbf, w2b, b2, x1, beta, (float*)d_out, nullptr, N, 256, 1024);
}
